// guide_upsample_83691732730054
// MI455X (gfx1250) — compile-verified, matrix operand order checked by reading, not run on hardware
//
#include <hip/hip_runtime.h>
#include <hip/hip_bf16.h>

// ---------------------------------------------------------------------------
// Types for CDNA5 WMMA
// ---------------------------------------------------------------------------
typedef _Float16 v16h __attribute__((ext_vector_type(16)));
typedef _Float16 v8h  __attribute__((ext_vector_type(8)));
typedef _Float16 v4h  __attribute__((ext_vector_type(4)));
typedef float    v8f  __attribute__((ext_vector_type(8)));

static __device__ __forceinline__ v16h cat8(v8h lo, v8h hi) {
    return __builtin_shufflevector(lo, hi, 0,1,2,3,4,5,6,7,8,9,10,11,12,13,14,15);
}

// ---------------------------------------------------------------------------
// Problem constants
// ---------------------------------------------------------------------------
#define DMODEL 512
#define NHEAD  8
#define DHEAD  64
#define NTOK   16384   // 128*128 guide tokens
#define LTOK   1024    // 32*32 low tokens
#define LDSTRIDE 40    // padded LDS row stride in halves (80B, keeps b128 runs aligned)

// ---------------------------------------------------------------------------
// WMMA GEMM:  C(MxN) = A(MxK) * B(KxN),  M = K = 512 fixed, N is a template
// constant so all staged accesses are base + immediate-offset (no addr chains).
//   A: f32 row-major (512x512 weight) -> converted to f16 through registers.
//   B: if B_TOK:  f16 token-major (B[n*512+k]) -> staged with
//                 global_load_async_to_lds_b128 (ASYNCcnt, no VGPR round trip)
//      else:      f32 d-major (B[k*N+n]) -> registers + cvt_pk + ds_store_b64
//   C: if C_TOK:  token-major C[n*512+m]  (Q/K/V)   else d-major C[m*N+n]
// Block: 256 threads = 8 waves; tile 128(M) x 64(N); K stepped by 32.
// Double-buffered LDS, one barrier per K-step. sched_barrier(0) pins the
// next-tile global loads before the WMMA section so their latency is covered.
// ---------------------------------------------------------------------------
template<typename BT, bool B_TOK, bool C_TOK, int N>
__global__ __launch_bounds__(256)
void gemm512_wmma_kernel(const float* __restrict__ A,
                         const BT*   __restrict__ B,
                         float*      __restrict__ C)
{
    __shared__ _Float16 As[2][128 * LDSTRIDE];
    __shared__ _Float16 Bs[2][64  * LDSTRIDE];

    const int tid  = threadIdx.x;
    const int wave = tid >> 5;
    const int lane = tid & 31;
    const int m0   = blockIdx.y * 128;
    const int n0   = blockIdx.x * 64;

    // ---- staging thread maps ----
    const int ac4 = (tid & 7) * 4;     // A: 4 consecutive k per thread
    const int ar0 = tid >> 3;          // A: row 0..31, 4 passes of 32 rows
    const int bc  = tid & 63;          // B d-major: column
    const int brg = (tid >> 6) * 4;    // B d-major: 4-row group base
    const int an  = tid >> 2;          // B async: token row 0..63
    const int aq  = (tid & 3) * 8;     // B async: 8-half (16B) chunk in k

    float4 areg[4];
    float  bregf[2][4];

    auto issue_loads = [&](int kk, int nbuf) {
        const float* abase = A + (size_t)(m0 + ar0) * DMODEL + kk + ac4;
#pragma unroll
        for (int p = 0; p < 4; ++p)
            areg[p] = *(const float4*)(abase + p * 32 * DMODEL);   // imm offsets
        if constexpr (B_TOK) {
            // async copy of the whole 32x64-half tile straight into LDS
            const unsigned lds_off =
                (unsigned)(size_t)&Bs[nbuf][an * LDSTRIDE + aq];
            const unsigned long long gaddr =
                (unsigned long long)(size_t)&B[(size_t)(n0 + an) * DMODEL + kk + aq];
            asm volatile("global_load_async_to_lds_b128 %0, %1, off"
                         :: "v"(lds_off), "v"(gaddr) : "memory");
        } else {
            const BT* bbase = B + (size_t)(kk + brg) * N + n0 + bc;
#pragma unroll
            for (int p = 0; p < 2; ++p)
#pragma unroll
                for (int j = 0; j < 4; ++j)
                    bregf[p][j] = bbase[(size_t)(p * 16 + j) * N]; // imm offsets
        }
    };

    auto store_tile = [&](int buf) {
#pragma unroll
        for (int p = 0; p < 4; ++p) {
            v4h h = { (_Float16)areg[p].x, (_Float16)areg[p].y,
                      (_Float16)areg[p].z, (_Float16)areg[p].w };
            *(v4h*)&As[buf][(ar0 + p * 32) * LDSTRIDE + ac4] = h;   // ds_store_b64
        }
        if constexpr (!B_TOK) {
#pragma unroll
            for (int p = 0; p < 2; ++p) {
                v4h h = { (_Float16)bregf[p][0], (_Float16)bregf[p][1],
                          (_Float16)bregf[p][2], (_Float16)bregf[p][3] };
                *(v4h*)&Bs[buf][bc * LDSTRIDE + p * 16 + brg] = h;  // contiguous in k
            }
        }
    };

    v8f acc[4];
#pragma unroll
    for (int t = 0; t < 4; ++t)
#pragma unroll
        for (int e = 0; e < 8; ++e) acc[t][e] = 0.0f;

    const int half = lane >> 4;        // K-group select per ISA fragment layout
    const int l16  = lane & 15;
    const int arow = wave * 16 + l16;

    issue_loads(0, 0);
    store_tile(0);
    if constexpr (B_TOK)
        asm volatile("s_wait_asynccnt 0x0" ::: "memory");
    __syncthreads();

    for (int step = 0; step < 16; ++step) {
        const int cur = step & 1;
        if (step < 15) issue_loads((step + 1) * 32, (step + 1) & 1);
        __builtin_amdgcn_sched_barrier(0);   // keep next-tile loads in flight here

        const _Float16* Ab = As[cur];
        const _Float16* Bb = Bs[cur];
        v8h a_lo = *(const v8h*)&Ab[arow * LDSTRIDE + half * 8];
        v8h a_hi = *(const v8h*)&Ab[arow * LDSTRIDE + 16 + half * 8];
        v16h afrag = cat8(a_lo, a_hi);

#pragma unroll
        for (int t = 0; t < 4; ++t) {
            const int brow = t * 16 + l16;
            v8h b_lo = *(const v8h*)&Bb[brow * LDSTRIDE + half * 8];
            v8h b_hi = *(const v8h*)&Bb[brow * LDSTRIDE + 16 + half * 8];
            acc[t] = __builtin_amdgcn_wmma_f32_16x16x32_f16(
                false, afrag, false, cat8(b_lo, b_hi), (short)0, acc[t], false, false);
        }

        if (step < 15) {
            store_tile((step + 1) & 1);        // other buffer: safe pre-barrier
            if constexpr (B_TOK)
                asm volatile("s_wait_asynccnt 0x0" ::: "memory");
        }
        __syncthreads();
    }

    // ---- store: element e of acc[t] is (M = m0+wave*16+half*8+e, n = n0+t*16+l16)
    const int Mbase = m0 + wave * 16 + half * 8;
#pragma unroll
    for (int t = 0; t < 4; ++t) {
        const int n = n0 + t * 16 + l16;
        if constexpr (C_TOK) {
            float4 lo = make_float4(acc[t][0], acc[t][1], acc[t][2], acc[t][3]);
            float4 hi = make_float4(acc[t][4], acc[t][5], acc[t][6], acc[t][7]);
            float4* p = (float4*)&C[(size_t)n * DMODEL + Mbase];
            p[0] = lo;
            p[1] = hi;
        } else {
            float* cb = &C[(size_t)Mbase * N + n];
#pragma unroll
            for (int e = 0; e < 8; ++e)
                cb[(size_t)e * N] = acc[t][e];                     // imm offsets
        }
    }
}

// ---------------------------------------------------------------------------
// Windowed attention: one wave per (token, head). Lane owns dims {2*lane,2*lane+1}
// of dm=64. 9 keys = 3x3 window on the 32x32 low grid. Scores via wave-wide
// shfl_xor reduction; softmax; weighted V sum; msg stored f16 token-major.
// ---------------------------------------------------------------------------
__global__ __launch_bounds__(256)
void attn_win9_kernel(const float* __restrict__ Q,    // [NTOK][512] token-major
                      const float* __restrict__ Kt,   // [LTOK][512] token-major
                      const float* __restrict__ Vt,   // [LTOK][512] token-major
                      _Float16*    __restrict__ msg)  // [NTOK][512] token-major
{
    const int lane = threadIdx.x & 31;
    const int wave = threadIdx.x >> 5;
    const int pair = blockIdx.x * 8 + wave;      // token*8 + head
    const int n    = pair >> 3;
    const int head = pair & 7;

    const int y  = n >> 7, x = n & 127;
    const int y0 = y >> 2, x0 = x >> 2;          // arange(128)*32//128

    const float2 q2 = *(const float2*)&Q[(size_t)n * DMODEL + head * DHEAD + lane * 2];

    float sc[9];
    int   midx[9];
#pragma unroll
    for (int k = 0; k < 9; ++k) {
        const int oy = k / 3 - 1, ox = k % 3 - 1;
        const int yy = min(max(y0 + oy, 0), 31);
        const int xx = min(max(x0 + ox, 0), 31);
        const int m  = yy * 32 + xx;
        midx[k] = m;
        const float2 k2 = *(const float2*)&Kt[(size_t)m * DMODEL + head * DHEAD + lane * 2];
        float p = q2.x * k2.x + q2.y * k2.y;
#pragma unroll
        for (int off = 16; off > 0; off >>= 1)
            p += __shfl_xor(p, off, 32);
        sc[k] = p * 0.125f;                      // 1/sqrt(64)
    }

    float mx = sc[0];
#pragma unroll
    for (int k = 1; k < 9; ++k) mx = fmaxf(mx, sc[k]);
    float w[9], s = 0.0f;
#pragma unroll
    for (int k = 0; k < 9; ++k) { w[k] = __expf(sc[k] - mx); s += w[k]; }
    const float inv = 1.0f / s;

    float ax = 0.0f, ab = 0.0f;
#pragma unroll
    for (int k = 0; k < 9; ++k) {
        const float2 v2 = *(const float2*)&Vt[(size_t)midx[k] * DMODEL + head * DHEAD + lane * 2];
        ax += w[k] * v2.x;
        ab += w[k] * v2.y;
    }

    _Float16* mp = msg + (size_t)n * DMODEL + head * DHEAD + lane * 2;
    mp[0] = (_Float16)(ax * inv);
    mp[1] = (_Float16)(ab * inv);
}

// ---------------------------------------------------------------------------
// In-place LayerNorm over d=512 per token; Y is d-major [512][NTOK] (= output
// layout (1,512,128,128)). Lane handles one token -> fully coalesced columns.
// ---------------------------------------------------------------------------
__global__ __launch_bounds__(256)
void layernorm_dmajor_kernel(float* __restrict__ Y,
                             const float* __restrict__ gamma,
                             const float* __restrict__ beta)
{
    const int t = blockIdx.x * blockDim.x + threadIdx.x;   // token 0..16383
    float s = 0.0f, s2 = 0.0f;
#pragma unroll 4
    for (int m = 0; m < DMODEL; ++m) {
        const float v = Y[(size_t)m * NTOK + t];
        s += v; s2 += v * v;
    }
    const float mean = s * (1.0f / DMODEL);
    const float var  = s2 * (1.0f / DMODEL) - mean * mean;
    const float inv  = rsqrtf(var + 1e-5f);
#pragma unroll 4
    for (int m = 0; m < DMODEL; ++m) {
        const size_t idx = (size_t)m * NTOK + t;
        Y[idx] = (Y[idx] - mean) * inv * gamma[m] + beta[m];
    }
}

// ---------------------------------------------------------------------------
// Launcher
// inputs: 0 low_feats(512*1024) 1 guide_feats(512*16384) 2 Wq 3 Wk 4 Wv 5 Wm
//         6 gamma(512) 7 beta(512).  Output: float32 (1,512,128,128).
// Workspace: Q 32MB + Kt 2MB + Vt 2MB + msg(f16) 16MB  ~= 52.5 MB.
// ---------------------------------------------------------------------------
extern "C" void kernel_launch(void* const* d_in, const int* in_sizes, int n_in,
                              void* d_out, int out_size, void* d_ws, size_t ws_size,
                              hipStream_t stream)
{
    const float* low   = (const float*)d_in[0];
    const float* guide = (const float*)d_in[1];
    const float* Wq    = (const float*)d_in[2];
    const float* Wk    = (const float*)d_in[3];
    const float* Wv    = (const float*)d_in[4];
    const float* Wm    = (const float*)d_in[5];
    const float* gamma = (const float*)d_in[6];
    const float* beta  = (const float*)d_in[7];
    float*       out   = (float*)d_out;

    float*    Q   = (float*)d_ws;                       // [NTOK][512]
    float*    Kt  = Q  + (size_t)NTOK * DMODEL;         // [LTOK][512]
    float*    Vt  = Kt + (size_t)LTOK * DMODEL;         // [LTOK][512]
    _Float16* msg = (_Float16*)(Vt + (size_t)LTOK * DMODEL); // [NTOK][512]

    const dim3 blk(256);

    // 1) Q = (Wq @ guide)^T, token-major
    gemm512_wmma_kernel<float, false, true, NTOK>
        <<<dim3(NTOK / 64, DMODEL / 128), blk, 0, stream>>>(Wq, guide, Q);
    // 2,3) K/V projections, token-major
    gemm512_wmma_kernel<float, false, true, LTOK>
        <<<dim3(LTOK / 64, DMODEL / 128), blk, 0, stream>>>(Wk, low, Kt);
    gemm512_wmma_kernel<float, false, true, LTOK>
        <<<dim3(LTOK / 64, DMODEL / 128), blk, 0, stream>>>(Wv, low, Vt);
    // 4) 3x3 windowed attention -> msg (f16, token-major)
    attn_win9_kernel<<<dim3((NTOK * NHEAD) / 8), blk, 0, stream>>>(Q, Kt, Vt, msg);
    // 5) out^T = Wm @ msg^T, d-major directly into d_out (async-to-LDS B path)
    gemm512_wmma_kernel<_Float16, true, false, NTOK>
        <<<dim3(NTOK / 64, DMODEL / 128), blk, 0, stream>>>(Wm, msg, out);
    // 6) in-place LayerNorm per token
    layernorm_dmajor_kernel<<<dim3(NTOK / 256), blk, 0, stream>>>(out, gamma, beta);
}
